// GraphSAGERouting2Hop_32229434589313
// MI455X (gfx1250) — matrix operand order repaired and checked by reading, not compiled, not run on hardware
//
#include <hip/hip_runtime.h>
#include <stdint.h>

typedef __attribute__((ext_vector_type(16))) __bf16 v16bf;
typedef __attribute__((ext_vector_type(8)))  __bf16 v8bf;
typedef __attribute__((ext_vector_type(8)))  float  v8f;

// ---------------------------------------------------------------------------
// Elementwise / graph kernels
// ---------------------------------------------------------------------------

__global__ void zero_f32_kernel(float* __restrict__ p, long long n) {
  long long stride = (long long)gridDim.x * blockDim.x;
  for (long long i = (long long)blockIdx.x * blockDim.x + threadIdx.x; i < n; i += stride)
    p[i] = 0.0f;
}

__global__ void degree_kernel(const int* __restrict__ dst, float* __restrict__ deg, int E) {
  int e = blockIdx.x * blockDim.x + threadIdx.x;
  if (e < E) unsafeAtomicAdd(&deg[dst[e]], 1.0f);
}

// One thread per (edge, 4-feature chunk); consecutive threads cover consecutive
// chunks of the same edge -> coalesced float4 gathers, fp32 L2 atomics scatter.
__global__ void scatter_add_kernel(const float* __restrict__ feat,
                                   const int* __restrict__ src,
                                   const int* __restrict__ dst,
                                   float* __restrict__ agg,
                                   int E, int D) {
  int nChunk = D >> 2;
  long long total = (long long)E * nChunk;
  long long gid = (long long)blockIdx.x * blockDim.x + threadIdx.x;
  if (gid >= total) return;
  int e = (int)(gid / nChunk);
  int c = (int)(gid % nChunk);
  int s = src[e], d = dst[e];
  const float4 v = *(const float4*)(feat + (long long)s * D + c * 4);
  float* out = agg + (long long)d * D + c * 4;
  unsafeAtomicAdd(out + 0, v.x);
  unsafeAtomicAdd(out + 1, v.y);
  unsafeAtomicAdd(out + 2, v.z);
  unsafeAtomicAdd(out + 3, v.w);
}

// A[i][k] = k < D ? agg[i][k] / max(deg[i],1) : self[i][k-D]     (bf16 output)
__global__ void build_A_kernel(const float* __restrict__ agg,
                               const float* __restrict__ self,
                               const float* __restrict__ deg,
                               __bf16* __restrict__ A,
                               long long N, int D) {
  int K = 2 * D;
  long long total = N * (long long)K;
  long long stride = (long long)gridDim.x * blockDim.x;
  for (long long i = (long long)blockIdx.x * blockDim.x + threadIdx.x; i < total; i += stride) {
    long long row = i / K;
    int k = (int)(i % K);
    float v;
    if (k < D) {
      float inv = 1.0f / fmaxf(deg[row], 1.0f);
      v = agg[row * D + k] * inv;
    } else {
      v = self[row * D + (k - D)];
    }
    A[i] = (__bf16)v;
  }
}

// Bt[n][k] = k < Din ? Wl[k][n] : Wr[k-Din][n]   (bf16, B transposed so WMMA
// B-fragment loads are contiguous per output column n)
__global__ void prep_Bt_kernel(const float* __restrict__ Wl,
                               const float* __restrict__ Wr,
                               __bf16* __restrict__ Bt,
                               int Din, int Dout) {
  int K = 2 * Din;
  int total = Dout * K;
  int stride = gridDim.x * blockDim.x;
  for (int i = blockIdx.x * blockDim.x + threadIdx.x; i < total; i += stride) {
    int n = i / K;
    int k = i % K;
    float v = (k < Din) ? Wl[(long long)k * Dout + n]
                        : Wr[(long long)(k - Din) * Dout + n];
    Bt[i] = (__bf16)v;
  }
}

// ---------------------------------------------------------------------------
// Fragment loaders (layouts per CDNA5 ISA 7.12.2, 16-bit A 16x32 / B 32x16)
// ---------------------------------------------------------------------------
__device__ __forceinline__ v16bf cat8(v8bf lo, v8bf hi) {
  return __builtin_shufflevector(lo, hi,
      0,1,2,3,4,5,6,7,8,9,10,11,12,13,14,15);
}
// A fragment: lane half h holds K = kb+h*8..+7 (VGPR0-3) and kb+16+h*8..+7 (VGPR4-7)
__device__ __forceinline__ v16bf loadAfrag(const __bf16* row, int kb, int half) {
  v8bf lo = *(const v8bf*)(row + kb + half * 8);
  v8bf hi = *(const v8bf*)(row + kb + 16 + half * 8);
  return cat8(lo, hi);
}
// B fragment: lane half h holds K = kb+h*8..+7 (VGPR0-3) and kb+16+h*8..+7 (VGPR4-7)
__device__ __forceinline__ v16bf loadBfrag(const __bf16* row, int kb, int half) {
  v8bf lo = *(const v8bf*)(row + kb + half * 8);
  v8bf hi = *(const v8bf*)(row + kb + 16 + half * 8);
  return cat8(lo, hi);
}

// ---------------------------------------------------------------------------
// bf16 WMMA GEMM:  C[M,Nc] = A[M,K] @ Bt[Nc,K]^T + bias, optional ReLU
// Block = 256 threads = 8 waves laid out 4(rows) x 2(cols);
// each wave computes 32 rows x 64 cols = 2x4 accumulator tiles.
// K is a template constant (256 / 512): the K-loop is fully unrolled so the
// double-buffer rotation is pure SSA renaming (no v_mov) and every fragment
// load folds kb into the 24-bit instruction offset (no per-step address VALU).
// ---------------------------------------------------------------------------
template <int K>
__global__ void __launch_bounds__(256)
sage_gemm_bf16_kernel(const __bf16* __restrict__ A,
                      const __bf16* __restrict__ Bt,
                      const float* __restrict__ bias,
                      float* __restrict__ C,
                      int M, int Nc, int relu) {
  const int lane = threadIdx.x & 31;
  const int wave = threadIdx.x >> 5;
  const int half = lane >> 4;      // 0/1 lane-half
  const int lmod = lane & 15;

  const int rowBase = blockIdx.x * 128 + (wave >> 1) * 32;
  const int colBase = blockIdx.y * 128 + (wave & 1) * 64;

  int m0 = rowBase + lmod;        if (m0 >= M) m0 = M - 1;  // clamp: keep EXEC all-ones
  int m1 = rowBase + 16 + lmod;   if (m1 >= M) m1 = M - 1;
  const __bf16* Arow0 = A + (size_t)m0 * K;
  const __bf16* Arow1 = A + (size_t)m1 * K;
  const __bf16* Brow0 = Bt + (size_t)(colBase + 0  + lmod) * K;
  const __bf16* Brow1 = Bt + (size_t)(colBase + 16 + lmod) * K;
  const __bf16* Brow2 = Bt + (size_t)(colBase + 32 + lmod) * K;
  const __bf16* Brow3 = Bt + (size_t)(colBase + 48 + lmod) * K;

  v8f acc[2][4] = {};

  constexpr int STEPS = K / 32;

  // prologue: fragments for step 0
  v16bf aC[2], bC[4];
  aC[0] = loadAfrag(Arow0, 0, half);
  aC[1] = loadAfrag(Arow1, 0, half);
  bC[0] = loadBfrag(Brow0, 0, half);
  bC[1] = loadBfrag(Brow1, 0, half);
  bC[2] = loadBfrag(Brow2, 0, half);
  bC[3] = loadBfrag(Brow3, 0, half);

#pragma unroll
  for (int s = 1; s < STEPS; ++s) {
    const int kb = s * 32;
    // prefetch step s fragments (constant offsets after full unroll)
    v16bf aN[2], bN[4];
    aN[0] = loadAfrag(Arow0, kb, half);
    aN[1] = loadAfrag(Arow1, kb, half);
    bN[0] = loadBfrag(Brow0, kb, half);
    bN[1] = loadBfrag(Brow1, kb, half);
    bN[2] = loadBfrag(Brow2, kb, half);
    bN[3] = loadBfrag(Brow3, kb, half);
    // consume step s-1 fragments
#pragma unroll
    for (int i = 0; i < 2; ++i)
#pragma unroll
      for (int j = 0; j < 4; ++j)
        acc[i][j] = __builtin_amdgcn_wmma_f32_16x16x32_bf16(
            false, (i ? aC[1] : aC[0]), false, bC[j],
            (short)0, acc[i][j], false, false);
    aC[0] = aN[0]; aC[1] = aN[1];
    bC[0] = bN[0]; bC[1] = bN[1]; bC[2] = bN[2]; bC[3] = bN[3];
  }
  // epilogue: last step
#pragma unroll
  for (int i = 0; i < 2; ++i)
#pragma unroll
    for (int j = 0; j < 4; ++j)
      acc[i][j] = __builtin_amdgcn_wmma_f32_16x16x32_bf16(
          false, (i ? aC[1] : aC[0]), false, bC[j],
          (short)0, acc[i][j], false, false);

  // C/D layout: col = colBase + j*16 + lane%16 ; row = rowBase + i*16 + half*8 + r
#pragma unroll
  for (int j = 0; j < 4; ++j) {
    int col = colBase + j * 16 + lmod;
    float bv = bias[col];
#pragma unroll
    for (int i = 0; i < 2; ++i) {
#pragma unroll
      for (int r = 0; r < 8; ++r) {
        int row = rowBase + i * 16 + half * 8 + r;
        if (row < M) {
          float v = acc[i][j][r] + bv;
          if (relu) v = fmaxf(v, 0.0f);
          C[(size_t)row * Nc + col] = v;
        }
      }
    }
  }
}

// ---------------------------------------------------------------------------
// Launch
// ---------------------------------------------------------------------------
extern "C" void kernel_launch(void* const* d_in, const int* in_sizes, int n_in,
                              void* d_out, int out_size, void* d_ws, size_t ws_size,
                              hipStream_t stream) {
  (void)n_in; (void)out_size; (void)ws_size;
  const float* x   = (const float*)d_in[0];
  const int*   ei  = (const int*)d_in[1];
  const float* W1l = (const float*)d_in[2];
  const float* b1  = (const float*)d_in[3];
  const float* W1r = (const float*)d_in[4];
  const float* W2l = (const float*)d_in[5];
  const float* b2  = (const float*)d_in[6];
  const float* W2r = (const float*)d_in[7];

  const int IN = 128, HID = 256;
  const long long N = in_sizes[0] / IN;
  const int E = in_sizes[1] / 2;
  const int* src  = ei;
  const int* dstI = ei + E;

  // Workspace carve-up (256B aligned). A-buffer aliased for both layers.
  auto alignUp = [](uintptr_t p) { return (p + 255) & ~(uintptr_t)255; };
  uintptr_t p = alignUp((uintptr_t)d_ws);
  float*  deg  = (float*)p;   p = alignUp(p + (size_t)N * sizeof(float));
  float*  agg  = (float*)p;   p = alignUp(p + (size_t)N * HID * sizeof(float));
  float*  h    = (float*)p;   p = alignUp(p + (size_t)N * HID * sizeof(float));
  __bf16* Abuf = (__bf16*)p;  p = alignUp(p + (size_t)N * 2 * HID * sizeof(__bf16));
  __bf16* Bt1  = (__bf16*)p;  p = alignUp(p + (size_t)2 * IN * HID * sizeof(__bf16));
  __bf16* Bt2  = (__bf16*)p;

  const int TB = 256;

  // ---- Layer 1: h = relu(mean_agg(x) @ W1_l + b1 + x @ W1_r) ----
  zero_f32_kernel<<<2048, TB, 0, stream>>>(deg, N);
  zero_f32_kernel<<<2048, TB, 0, stream>>>(agg, N * (long long)IN);
  degree_kernel<<<(E + TB - 1) / TB, TB, 0, stream>>>(dstI, deg, E);
  {
    long long total = (long long)E * (IN / 4);
    scatter_add_kernel<<<(unsigned)((total + TB - 1) / TB), TB, 0, stream>>>(
        x, src, dstI, agg, E, IN);
  }
  prep_Bt_kernel<<<256, TB, 0, stream>>>(W1l, W1r, Bt1, IN, HID);
  build_A_kernel<<<4096, TB, 0, stream>>>(agg, x, deg, Abuf, N, IN);
  {
    dim3 grid((unsigned)((N + 127) / 128), HID / 128);
    sage_gemm_bf16_kernel<256><<<grid, TB, 0, stream>>>(
        Abuf, Bt1, b1, h, (int)N, HID, /*relu=*/1);
  }

  // ---- Layer 2: out = mean_agg(h) @ W2_l + b2 + h @ W2_r ----
  zero_f32_kernel<<<2048, TB, 0, stream>>>(agg, N * (long long)HID);
  {
    long long total = (long long)E * (HID / 4);
    scatter_add_kernel<<<(unsigned)((total + TB - 1) / TB), TB, 0, stream>>>(
        h, src, dstI, agg, E, HID);
  }
  prep_Bt_kernel<<<256, TB, 0, stream>>>(W2l, W2r, Bt2, HID, HID);
  build_A_kernel<<<4096, TB, 0, stream>>>(agg, h, deg, Abuf, N, HID);
  {
    dim3 grid((unsigned)((N + 127) / 128), HID / 128);
    sage_gemm_bf16_kernel<512><<<grid, TB, 0, stream>>>(
        Abuf, Bt2, b2, (float*)d_out, (int)N, HID, /*relu=*/0);
  }
}
